// PointNet2Classifier_88880053223795
// MI455X (gfx1250) — matrix operand order repaired and checked by reading, not compiled, not run on hardware
//
#include <hip/hip_runtime.h>
#include <hip/hip_bf16.h>

typedef _Float16 v16h __attribute__((ext_vector_type(16)));
typedef _Float16 v8h  __attribute__((ext_vector_type(8)));
typedef float    v8f  __attribute__((ext_vector_type(8)));

static constexpr int BATCH = 16;
static constexpr int NPTS0 = 8192;

// ---------------------------------------------------------------- utilities
__device__ __forceinline__ unsigned ballot32(bool p) {
#if __has_builtin(__builtin_amdgcn_ballot_w32)
  return (unsigned)__builtin_amdgcn_ballot_w32(p);
#else
  return (unsigned)__ballot(p);
#endif
}

// wave32 argmax butterfly via ds_swizzle (group-of-32 xor mode, no LDS memory touched)
__device__ __forceinline__ void wave_argmax(float& bm, int& ba) {
#define AM_STEP(XM)                                                                         \
  {                                                                                         \
    float om = __int_as_float(__builtin_amdgcn_ds_swizzle(__float_as_int(bm),               \
                                                          (((XM) << 10) | 31)));            \
    int oa = __builtin_amdgcn_ds_swizzle(ba, (((XM) << 10) | 31));                          \
    if (om > bm || (om == bm && oa < ba)) { bm = om; ba = oa; }                             \
  }
  AM_STEP(16) AM_STEP(8) AM_STEP(4) AM_STEP(2) AM_STEP(1)
#undef AM_STEP
}

// WMMA fragment loaders (16-bit A/B layouts per CDNA5 ISA 7.12.2)
__device__ __forceinline__ v16h load_afrag(const _Float16* p) {  // p = &Arow[kb + hi*8]
  union { v16h v; v8h h[2]; } u;
  u.h[0] = *(const v8h*)(p);        // K = kb + hi*8 .. +7
  u.h[1] = *(const v8h*)(p + 16);   // K = kb + 16 + hi*8 .. +7
  return u.v;
}
__device__ __forceinline__ v16h load_bfrag(const _Float16* p) {  // p = &Brow[kb + hi*8]
  union { v16h v; v8h h[2]; } u;
  u.h[0] = *(const v8h*)(p);        // K = kb + hi*8 .. +7
  u.h[1] = *(const v8h*)(p + 16);   // K = kb + 16 + hi*8 .. +7
  return u.v;
}

// ---------------------------------------------------------------- transpose (B,3,N) -> (B,N,3)
__global__ void k_transpose(const float* __restrict__ x, float* __restrict__ xyz) {
  int i = blockIdx.x * 256 + threadIdx.x;  // over BATCH*NPTS0
  if (i >= BATCH * NPTS0) return;
  int b = i / NPTS0, n = i % NPTS0;
  xyz[(size_t)i * 3 + 0] = x[((size_t)b * 3 + 0) * NPTS0 + n];
  xyz[(size_t)i * 3 + 1] = x[((size_t)b * 3 + 1) * NPTS0 + n];
  xyz[(size_t)i * 3 + 2] = x[((size_t)b * 3 + 2) * NPTS0 + n];
}

// ---------------------------------------------------------------- farthest point sampling
// One workgroup per batch; dist[] in LDS; argmax via ds_swizzle butterfly + 1 LDS hop.
template <int NP>
__global__ __launch_bounds__(256) void k_fps(const float* __restrict__ xyz,
                                             int* __restrict__ outidx, int npoint) {
  __shared__ float dist[NP];
  __shared__ float rmax[8];
  __shared__ int   rarg[8];
  __shared__ int   sfar;
  int tid = threadIdx.x;
  int wave = tid >> 5, lane = tid & 31;
  const float* p = xyz + (size_t)blockIdx.x * NP * 3;
  for (int j = tid; j < NP; j += 256) dist[j] = 1e10f;
  if (tid == 0) sfar = 0;
  __syncthreads();
  for (int it = 0; it < npoint; ++it) {
    int far = sfar;
    if (tid == 0) outidx[blockIdx.x * npoint + it] = far;
    float fx = p[far * 3 + 0], fy = p[far * 3 + 1], fz = p[far * 3 + 2];
    float bm = -1.0f; int ba = 0x7fffffff;
    for (int j = tid; j < NP; j += 256) {
      float dx = p[j * 3 + 0] - fx, dy = p[j * 3 + 1] - fy, dz = p[j * 3 + 2] - fz;
      float d = dx * dx + dy * dy + dz * dz;
      float dd = dist[j];
      if (d < dd) { dd = d; dist[j] = d; }
      if (dd > bm) { bm = dd; ba = j; }  // strided order keeps lowest j per thread on ties
    }
    wave_argmax(bm, ba);
    if (lane == 0) { rmax[wave] = bm; rarg[wave] = ba; }
    __syncthreads();
    if (wave == 0) {
      float m2 = (lane < 8) ? rmax[lane] : -1.0f;
      int   a2 = (lane < 8) ? rarg[lane] : 0x7fffffff;
      wave_argmax(m2, a2);
      if (lane == 0) sfar = a2;
    }
    __syncthreads();
  }
}

// ---------------------------------------------------------------- ball query + group
// One wave (32 threads) per query point; ordered compaction via ballot+prefix.
__global__ void k_group(const float* __restrict__ xyz, const int* __restrict__ fpsidx,
                        const _Float16* __restrict__ feats, float* __restrict__ new_xyz,
                        _Float16* __restrict__ out, int Npts, int Ssamp, int nsample,
                        int featC, int Cpad, float r2) {
  int q = blockIdx.x;  // over BATCH*Ssamp
  int b = q / Ssamp;
  int lane = threadIdx.x;  // 32 threads
  int center = fpsidx[q];
  const float* cptr = xyz + ((size_t)b * Npts + center) * 3;
  float cx = cptr[0], cy = cptr[1], cz = cptr[2];
  if (lane < 3) new_xyz[(size_t)q * 3 + lane] = cptr[lane];
  __shared__ float row0[3];
  __shared__ int   sidx0;
  int filled = 0;
  for (int base = 0; base < Npts && filled < nsample; base += 32) {
    int j = base + lane;
    bool ok = false;
    float rx = 0.f, ry = 0.f, rz = 0.f;
    if (j < Npts) {
      const float* pp = xyz + ((size_t)b * Npts + j) * 3;
      rx = pp[0] - cx; ry = pp[1] - cy; rz = pp[2] - cz;
      ok = (rx * rx + ry * ry + rz * rz) <= r2;
    }
    unsigned m = ballot32(ok);
    int prefix = __popc(m & ((1u << lane) - 1u));
    int slot = filled + prefix;
    if (ok && slot < nsample) {
      _Float16* o = out + ((size_t)q * nsample + slot) * Cpad;
      o[0] = (_Float16)rx; o[1] = (_Float16)ry; o[2] = (_Float16)rz;
      if (featC > 0) {
        const _Float16* f = feats + ((size_t)b * Npts + j) * featC;
        for (int c = 0; c < featC; ++c) o[3 + c] = f[c];
      }
      for (int c = 3 + featC; c < Cpad; ++c) o[c] = (_Float16)0.f;
      if (slot == 0) { row0[0] = rx; row0[1] = ry; row0[2] = rz; sidx0 = j; }
    }
    filled += __popc(m);
  }
  if (filled > nsample) filled = nsample;
  __syncthreads();
  for (int slot = filled + lane; slot < nsample; slot += 32) {
    _Float16* o = out + ((size_t)q * nsample + slot) * Cpad;
    o[0] = (_Float16)row0[0]; o[1] = (_Float16)row0[1]; o[2] = (_Float16)row0[2];
    if (featC > 0) {
      const _Float16* f = feats + ((size_t)b * Npts + sidx0) * featC;
      for (int c = 0; c < featC; ++c) o[3 + c] = f[c];
    }
    for (int c = 3 + featC; c < Cpad; ++c) o[c] = (_Float16)0.f;
  }
}

// ---------------------------------------------------------------- group-all (SA3): [xyz2(3)|p2(256)|pad]
__global__ void k_group_all(const float* __restrict__ xyz2, const _Float16* __restrict__ p2,
                            _Float16* __restrict__ g3, int rows, int featC, int Cpad) {
  int i = blockIdx.x * 256 + threadIdx.x;
  if (i >= rows) return;
  _Float16* o = g3 + (size_t)i * Cpad;
  o[0] = (_Float16)xyz2[(size_t)i * 3 + 0];
  o[1] = (_Float16)xyz2[(size_t)i * 3 + 1];
  o[2] = (_Float16)xyz2[(size_t)i * 3 + 2];
  const _Float16* f = p2 + (size_t)i * featC;
  for (int c = 0; c < featC; ++c) o[3 + c] = f[c];
  for (int c = 3 + featC; c < Cpad; ++c) o[c] = (_Float16)0.f;
}

// ---------------------------------------------------------------- weight prep: f32 W -> padded f16 + fused BN scale/shift
__global__ void k_prep(const float* __restrict__ W, const float* __restrict__ bias,
                       const float* __restrict__ g, const float* __restrict__ be,
                       const float* __restrict__ m, const float* __restrict__ v,
                       _Float16* __restrict__ Wh, float* __restrict__ s, float* __restrict__ t,
                       int Cout, int Cin, int Kpad, int Npad) {
  int i = blockIdx.x * 256 + threadIdx.x;
  if (i < Npad * Kpad) {
    int o = i / Kpad, k = i % Kpad;
    float w = (o < Cout && k < Cin) ? W[(size_t)o * Cin + k] : 0.f;
    Wh[i] = (_Float16)w;
  }
  if (i < Npad) {
    if (i < Cout) {
      float sc = g ? g[i] * rsqrtf(v[i] + 1e-5f) : 1.f;
      s[i] = sc;
      t[i] = g ? (bias[i] - m[i]) * sc + be[i] : bias[i];
    } else { s[i] = 0.f; t[i] = 0.f; }
  }
}

// ---------------------------------------------------------------- big WMMA GEMM
// Block = 8 waves; each wave owns a 64x32 output tile (4 M-tiles x 2 N-tiles).
// KPAD is a template parameter so the K loop fully unrolls: the two-stage
// fragment buffers rotate by SSA renaming (no v_mov copies, no hazard NOPs).
// Requires: M % 512 == 0, Npad % 32 == 0, KPAD % 32 == 0.
template <int KPAD>
__global__ __launch_bounds__(256) void k_gemm_big(const _Float16* __restrict__ A,
                                                  const _Float16* __restrict__ Wh,
                                                  const float* __restrict__ s,
                                                  const float* __restrict__ t,
                                                  _Float16* __restrict__ out16,
                                                  int Npad, int doRelu) {
  int tid = threadIdx.x;
  int wave = tid >> 5, lane = tid & 31;
  int hi = lane >> 4, r = lane & 15;
  size_t rowbase = ((size_t)blockIdx.x * 8 + wave) * 64;  // 4 tiles * 16 rows
  int ncol0 = blockIdx.y * 32;

  const _Float16* Ap = A + (rowbase + r) * KPAD + hi * 8;
  const _Float16* Bp = Wh + (size_t)(ncol0 + r) * KPAD + hi * 8;
  constexpr size_t STR = (size_t)16 * KPAD;  // rows between 16-row sub-tiles

  v8f acc[4][2];
#pragma unroll
  for (int i = 0; i < 4; ++i)
#pragma unroll
    for (int j = 0; j < 2; ++j) acc[i][j] = (v8f){0.f, 0.f, 0.f, 0.f, 0.f, 0.f, 0.f, 0.f};

  v16h aS[2][4], bS[2][2];
#pragma unroll
  for (int i = 0; i < 4; ++i) aS[0][i] = load_afrag(Ap + i * STR);
#pragma unroll
  for (int j = 0; j < 2; ++j) bS[0][j] = load_bfrag(Bp + j * STR);

#pragma unroll
  for (int kb = 0; kb < KPAD; kb += 32) {
    const int st = (kb >> 5) & 1;  // compile-time after unroll
    if (kb + 32 < KPAD) {
      __builtin_prefetch(Ap + kb + 64, 0, 3);  // warm K+2 (near-cache)
#pragma unroll
      for (int i = 0; i < 4; ++i) aS[st ^ 1][i] = load_afrag(Ap + i * STR + kb + 32);
#pragma unroll
      for (int j = 0; j < 2; ++j) bS[st ^ 1][j] = load_bfrag(Bp + j * STR + kb + 32);
    }
#pragma unroll
    for (int j = 0; j < 2; ++j)
#pragma unroll
      for (int i = 0; i < 4; ++i)
        acc[i][j] = __builtin_amdgcn_wmma_f32_16x16x32_f16(false, aS[st][i], false, bS[st][j],
                                                           (short)0, acc[i][j], false, false);
  }

#pragma unroll
  for (int j = 0; j < 2; ++j) {
    int col = ncol0 + j * 16 + r;
    float sc = s[col], sh = t[col];
#pragma unroll
    for (int i = 0; i < 4; ++i) {
#pragma unroll
      for (int e = 0; e < 8; ++e) {
        size_t row = rowbase + i * 16 + hi * 8 + e;
        float vv = acc[i][j][e] * sc + sh;
        if (doRelu) vv = vv > 0.f ? vv : 0.f;
        out16[row * Npad + col] = (_Float16)vv;
      }
    }
  }
}

// ---------------------------------------------------------------- small WMMA GEMM (M == 16, heads)
// One wave per block; one 16x16 tile per block; barrier-free.
__global__ __launch_bounds__(32) void k_gemm_small(const _Float16* __restrict__ A,
                                                   const _Float16* __restrict__ Wh,
                                                   const float* __restrict__ s,
                                                   const float* __restrict__ t,
                                                   _Float16* __restrict__ out16,
                                                   float* __restrict__ out32,
                                                   int Kpad, int Npad, int Nreal, int doRelu) {
  int lane = threadIdx.x;
  int hi = lane >> 4, r = lane & 15;
  int ncol0 = blockIdx.x * 16;
  const _Float16* Ap = A + (size_t)r * Kpad + hi * 8;
  const _Float16* Bp = Wh + (size_t)(ncol0 + r) * Kpad + hi * 8;
  v8f acc = {0.f, 0.f, 0.f, 0.f, 0.f, 0.f, 0.f, 0.f};
  for (int kb = 0; kb < Kpad; kb += 32) {
    v16h a = load_afrag(Ap + kb);
    v16h b = load_bfrag(Bp + kb);
    acc = __builtin_amdgcn_wmma_f32_16x16x32_f16(false, a, false, b, (short)0, acc, false, false);
  }
  int col = ncol0 + r;
  float sc = s[col], sh = t[col];
#pragma unroll
  for (int e = 0; e < 8; ++e) {
    int row = hi * 8 + e;  // M = 16
    float vv = acc[e] * sc + sh;
    if (doRelu) vv = vv > 0.f ? vv : 0.f;
    if (out32) {
      if (col < Nreal) out32[(size_t)row * Nreal + col] = vv;
    } else {
      out16[(size_t)row * Npad + col] = (_Float16)vv;
    }
  }
}

// ---------------------------------------------------------------- max-pool over neighborhood axis
__global__ void k_maxpool(const _Float16* __restrict__ in, _Float16* __restrict__ out,
                          int groups, int nsample, int C) {
  int i = blockIdx.x * 256 + threadIdx.x;
  if (i >= groups * C) return;
  int g = i / C, c = i % C;
  float m = -3.0e38f;
  for (int j = 0; j < nsample; ++j) {
    float v = (float)in[((size_t)g * nsample + j) * C + c];
    m = v > m ? v : m;
  }
  out[(size_t)g * C + c] = (_Float16)m;
}

// ---------------------------------------------------------------- host orchestration
struct LD { int wi, bi, gi, Cout, Cin, Kpad, Npad; };

extern "C" void kernel_launch(void* const* d_in, const int* in_sizes, int n_in,
                              void* d_out, int out_size, void* d_ws, size_t ws_size,
                              hipStream_t stream) {
  (void)in_sizes; (void)n_in; (void)out_size; (void)ws_size;
  const float* x = (const float*)d_in[0];

  // layer table: d_in index of W, bias, g (be=g+1, m=g+2, v=g+3); -1 => no BN
  static const LD Ltab[12] = {
    { 1,  2,  3,   64,    3,   32,   64},   // sa1 l0
    { 7,  8,  9,   64,   64,   64,   64},   // sa1 l1
    {13, 14, 15,  128,   64,   64,  128},   // sa1 l2
    {19, 20, 21,  128,  131,  160,  128},   // sa2 l0
    {25, 26, 27,  128,  128,  128,  128},   // sa2 l1
    {31, 32, 33,  256,  128,  128,  256},   // sa2 l2
    {37, 38, 39,  256,  259,  288,  256},   // sa3 l0
    {43, 44, 45,  512,  256,  256,  512},   // sa3 l1
    {49, 50, 51, 1024,  512,  512, 1024},   // sa3 l2
    {55, 56, 57,  512, 1024, 1024,  512},   // fc1+bn1
    {61, 62, 63,  256,  512,  512,  256},   // fc2+bn2
    {67, 68, -1,   40,  256,  256,   48},   // fc3
  };
  size_t whOff[12], stOff[12], whTot = 0, stTot = 0;
  for (int i = 0; i < 12; ++i) {
    whOff[i] = whTot; whTot += (size_t)Ltab[i].Npad * Ltab[i].Kpad;
    stOff[i] = stTot; stTot += (size_t)Ltab[i].Npad;
  }

  const int S1 = 512, NS1 = 32, S2 = 128, NS2 = 64;
  const size_t M1 = (size_t)BATCH * S1 * NS1;  // 262144
  const size_t M2 = (size_t)BATCH * S2 * NS2;  // 131072
  const size_t M3 = (size_t)BATCH * S2;        // 2048

  size_t off = 0;
  auto alloc = [&](size_t bytes) -> void* {
    void* p = (char*)d_ws + off;
    off += (bytes + 255) & ~(size_t)255;
    return p;
  };
  float*    xyz  = (float*)alloc(sizeof(float) * BATCH * NPTS0 * 3);
  int*      fps1 = (int*)alloc(sizeof(int) * BATCH * S1);
  float*    xyz1 = (float*)alloc(sizeof(float) * BATCH * S1 * 3);
  int*      fps2 = (int*)alloc(sizeof(int) * BATCH * S2);
  float*    xyz2 = (float*)alloc(sizeof(float) * BATCH * S2 * 3);
  _Float16* g1   = (_Float16*)alloc(sizeof(_Float16) * M1 * 32);
  _Float16* actA = (_Float16*)alloc(sizeof(_Float16) * M1 * 64);    // ping
  _Float16* actB = (_Float16*)alloc(sizeof(_Float16) * M1 * 64);    // pong
  _Float16* actC = (_Float16*)alloc(sizeof(_Float16) * M1 * 128);   // wide final-layer buf
  _Float16* p1   = (_Float16*)alloc(sizeof(_Float16) * BATCH * S1 * 128);
  _Float16* g2   = (_Float16*)alloc(sizeof(_Float16) * M2 * 160);
  _Float16* p2   = (_Float16*)alloc(sizeof(_Float16) * BATCH * S2 * 256);
  _Float16* g3   = (_Float16*)alloc(sizeof(_Float16) * M3 * 288);
  _Float16* p3   = (_Float16*)alloc(sizeof(_Float16) * BATCH * 1024);
  _Float16* h1   = (_Float16*)alloc(sizeof(_Float16) * BATCH * 512);
  _Float16* h2   = (_Float16*)alloc(sizeof(_Float16) * BATCH * 256);
  _Float16* Wh   = (_Float16*)alloc(sizeof(_Float16) * whTot);
  float*    sbuf = (float*)alloc(sizeof(float) * stTot);
  float*    tbuf = (float*)alloc(sizeof(float) * stTot);

  // ---- weight prep (all layers, independent)
  for (int i = 0; i < 12; ++i) {
    const LD& l = Ltab[i];
    const float* gp  = l.gi >= 0 ? (const float*)d_in[l.gi]     : nullptr;
    const float* bep = l.gi >= 0 ? (const float*)d_in[l.gi + 1] : nullptr;
    const float* mp  = l.gi >= 0 ? (const float*)d_in[l.gi + 2] : nullptr;
    const float* vp  = l.gi >= 0 ? (const float*)d_in[l.gi + 3] : nullptr;
    int tot = l.Npad * l.Kpad;
    k_prep<<<(tot + 255) / 256, 256, 0, stream>>>(
        (const float*)d_in[l.wi], (const float*)d_in[l.bi], gp, bep, mp, vp,
        Wh + whOff[i], sbuf + stOff[i], tbuf + stOff[i], l.Cout, l.Cin, l.Kpad, l.Npad);
  }

  auto gemm_big = [&](int li, const _Float16* Ain, _Float16* O16, size_t M, int relu) {
    const LD& l = Ltab[li];
    dim3 grid((unsigned)(M / 512), (unsigned)(l.Npad / 32));
    const _Float16* W = Wh + whOff[li];
    const float* sp = sbuf + stOff[li];
    const float* tp = tbuf + stOff[li];
    switch (l.Kpad) {
      case 32:  k_gemm_big<32><<<grid, 256, 0, stream>>>(Ain, W, sp, tp, O16, l.Npad, relu); break;
      case 64:  k_gemm_big<64><<<grid, 256, 0, stream>>>(Ain, W, sp, tp, O16, l.Npad, relu); break;
      case 128: k_gemm_big<128><<<grid, 256, 0, stream>>>(Ain, W, sp, tp, O16, l.Npad, relu); break;
      case 160: k_gemm_big<160><<<grid, 256, 0, stream>>>(Ain, W, sp, tp, O16, l.Npad, relu); break;
      case 256: k_gemm_big<256><<<grid, 256, 0, stream>>>(Ain, W, sp, tp, O16, l.Npad, relu); break;
      case 288: k_gemm_big<288><<<grid, 256, 0, stream>>>(Ain, W, sp, tp, O16, l.Npad, relu); break;
      case 512: k_gemm_big<512><<<grid, 256, 0, stream>>>(Ain, W, sp, tp, O16, l.Npad, relu); break;
      default: break;
    }
  };
  auto gemm_small = [&](int li, const _Float16* Ain, _Float16* O16, float* O32,
                        int Nreal, int relu) {
    const LD& l = Ltab[li];
    k_gemm_small<<<l.Npad / 16, 32, 0, stream>>>(Ain, Wh + whOff[li], sbuf + stOff[li],
                                                 tbuf + stOff[li], O16, O32,
                                                 l.Kpad, l.Npad, Nreal, relu);
  };

  // ---- stage 0: layout
  k_transpose<<<(BATCH * NPTS0 + 255) / 256, 256, 0, stream>>>(x, xyz);

  // ---- SA1
  k_fps<NPTS0><<<BATCH, 256, 0, stream>>>(xyz, fps1, S1);
  k_group<<<BATCH * S1, 32, 0, stream>>>(xyz, fps1, nullptr, xyz1, g1,
                                         NPTS0, S1, NS1, 0, 32, 0.04f);
  gemm_big(0, g1,   actA, M1, 1);
  gemm_big(1, actA, actB, M1, 1);
  gemm_big(2, actB, actC, M1, 1);
  k_maxpool<<<(BATCH * S1 * 128 + 255) / 256, 256, 0, stream>>>(actC, p1, BATCH * S1, NS1, 128);

  // ---- SA2
  k_fps<512><<<BATCH, 256, 0, stream>>>(xyz1, fps2, S2);
  k_group<<<BATCH * S2, 32, 0, stream>>>(xyz1, fps2, p1, xyz2, g2,
                                         S1, S2, NS2, 128, 160, 0.16f);
  gemm_big(3, g2,   actA, M2, 1);
  gemm_big(4, actA, actB, M2, 1);
  gemm_big(5, actB, actC, M2, 1);
  k_maxpool<<<(BATCH * S2 * 256 + 255) / 256, 256, 0, stream>>>(actC, p2, BATCH * S2, NS2, 256);

  // ---- SA3 (group all)
  k_group_all<<<((int)M3 + 255) / 256, 256, 0, stream>>>(xyz2, p2, g3, (int)M3, 256, 288);
  gemm_big(6, g3,   actA, M3, 1);
  gemm_big(7, actA, actB, M3, 1);
  gemm_big(8, actB, actC, M3, 1);
  k_maxpool<<<(BATCH * 1024 + 255) / 256, 256, 0, stream>>>(actC, p3, BATCH, S2, 1024);

  // ---- heads (M = 16 = one row-tile)
  gemm_small(9,  p3, h1, nullptr, 512, 1);
  gemm_small(10, h1, h2, nullptr, 256, 1);
  gemm_small(11, h2, nullptr, (float*)d_out, 40, 0);
}
